// CausalMultiheadAttentionoWithRope_42125039239931
// MI455X (gfx1250) — matrix operand order repaired and checked by reading, not compiled, not run on hardware
//
#include <hip/hip_runtime.h>
#include <hip/hip_fp16.h>

typedef __attribute__((ext_vector_type(16))) _Float16 v16h;
typedef __attribute__((ext_vector_type(8)))  _Float16 v8h;
typedef __attribute__((ext_vector_type(8)))  float    v8f;
typedef __attribute__((ext_vector_type(4)))  unsigned int v4u;
typedef __attribute__((ext_vector_type(8)))  int      v8i;
typedef __attribute__((ext_vector_type(4)))  int      v4i;

#define BATCH 4
#define SEQ   2048
#define DM    1024
#define NH    16
#define DH    64
#define QK_SCALE 0.125f        /* 1/sqrt(64), folded into Q */
#define NEG_INF (-1e30f)
/* ln(10000)/32 : inv_freq = exp(-pairk * ln(theta)/(d_h/2)) */
#define ROPE_LN 0.28782313662425583f

/* LDS B-panel: 64 rows x 32 f16 (64B data) + 16B TDM pad -> 80B row stride.
 * 16 lanes x b128 reads at 80B stride touch all 64 LDS banks exactly once. */
#define ROWB   80
#define PANELB (64 * ROWB)     /* 5120 B per buffer, x2 double-buffered */
#define KSTEPS (DM / 32)

union FragU { v16h v; v8h h[2]; _Float16 e[16]; };

/* A fragment 16x32 f16, contiguous k order used by this code: lanes 0-15 hold K=0..15, lanes 16-31 hold K=16..31 */
__device__ __forceinline__ v16h loadA(const _Float16* p, int hi) {
  FragU u;
  u.h[0] = *(const v8h*)(p + 16 * hi);
  u.h[1] = *(const v8h*)(p + 16 * hi + 8);
  return u.v;
}
/* B fragment 32x16 f16, contiguous k order used by this code: lanes 0-15 hold K=0..15, lanes 16-31 hold K=16..31 */
__device__ __forceinline__ v16h loadB(const _Float16* p, int hi) {
  FragU u;
  u.h[0] = *(const v8h*)(p + 16 * hi);
  u.h[1] = *(const v8h*)(p + 16 * hi + 8);
  return u.v;
}
/* B fragment from the padded LDS panel: row = local n, 32B contiguous per lane half */
__device__ __forceinline__ v16h ldsB(const char* panel, int row, int hi) {
  FragU u;
  const char* p = panel + row * ROWB + 32 * hi;
  u.h[0] = *(const v8h*)p;
  u.h[1] = *(const v8h*)(p + 16);
  return u.v;
}
__device__ __forceinline__ v8f wmma_f16(v16h a, v16h b, v8f c) {
  return __builtin_amdgcn_wmma_f32_16x16x32_f16(false, a, false, b, (short)0, c, false, false);
}

/* ---- Tensor Data Mover: DMA a 64-row x 32-f16 B panel (row stride DM) into LDS ----
 * D# group0: count=1 | lds_addr | global_addr | type=2
 * D# group1: data_size=1(2B), pad_enable, pad_interval=3(16 dw), pad_amount=3(4 dw),
 *            tensor_dim0=DM, tensor_dim1=ntot, tile_dim0=32, tile_dim1=64, dim0_stride=DM */
__device__ __forceinline__ void tdm_load_panel(unsigned lds_off, const _Float16* gptr, int ntot) {
  unsigned long long ga = (unsigned long long)(uintptr_t)gptr;
  v4u g0;
  g0.x = 1u;                                        /* count=1, user mode */
  g0.y = lds_off;                                   /* LDS byte address */
  g0.z = (unsigned)(ga & 0xffffffffu);
  g0.w = (unsigned)((ga >> 32) & 0x01ffffffu) | (2u << 30);   /* addr[56:32] | type=2 */
  v8i g1;
  g1[0] = (1 << 16) | (1 << 20) | (3 << 22) | (3 << 25);      /* dsz=2B, pad en, int=16dw, amt=4dw */
  g1[1] = (int)(((unsigned)DM & 0xffffu) << 16);              /* tensor_dim0 lo16 */
  g1[2] = (int)(((unsigned)DM >> 16) | (((unsigned)ntot & 0xffffu) << 16)); /* dim0 hi | dim1 lo */
  g1[3] = (int)(((unsigned)ntot >> 16) | (32u << 16));        /* dim1 hi | tile_dim0=32 */
  g1[4] = 64;                                                  /* tile_dim1=64, tile_dim2=0 */
  g1[5] = DM;                                                  /* tensor_dim0_stride lo32 */
  g1[6] = 0;
  g1[7] = 0;
  v4i z4 = {0, 0, 0, 0};
#if defined(__clang_major__) && __clang_major__ >= 23
  v8i z8 = {0, 0, 0, 0, 0, 0, 0, 0};
  __builtin_amdgcn_tensor_load_to_lds(g0, g1, z4, z4, z8, 0);
#else
  __builtin_amdgcn_tensor_load_to_lds(g0, g1, z4, z4, 0);
#endif
}

/* ---------------- fp32 -> f16 conversion ---------------- */
__global__ void cvt_f32_f16_kernel(const float* __restrict__ in, _Float16* __restrict__ out, int n) {
  int i = blockIdx.x * blockDim.x + threadIdx.x;
  if (i < n) out[i] = (_Float16)in[i];
}

/* ---------------- QKV GEMM + fused RoPE epilogue ----------------
 * qkv[m, e] = sum_d x[m, d] * Wqkv[e, d]
 * Block = 4 waves = 64x64 tile; B panel TDM-staged in LDS (double-buffered),
 * A fragments software-pipelined in registers, B fragments batch-loaded so the
 * ds waits stagger instead of serializing each WMMA.
 * grid: (8192/64, 3072/64), block 128. */
__global__ __launch_bounds__(128)
void qkv_rope_kernel(const _Float16* __restrict__ xh, const _Float16* __restrict__ wh,
                     _Float16* __restrict__ qh, _Float16* __restrict__ kh,
                     _Float16* __restrict__ vth)
{
  __shared__ char smem[2 * PANELB];
  const int tid = threadIdx.x;
  const int lane = tid & 31;
  const int llo = lane & 15, hi = lane >> 4;
  const int m0 = blockIdx.x * 64 + (tid >> 5) * 16;
  const int n0 = blockIdx.y * 64;
  const unsigned lds0 = (unsigned)(size_t)smem;

  const _Float16* arow = xh + (size_t)(m0 + llo) * DM;
  const _Float16* wrow = wh + (size_t)n0 * DM;

  v8f acc[4] = {};
  if (tid < 32) tdm_load_panel(lds0, wrow, 3 * DM);

  v16h a_cur = loadA(arow, hi);
  for (int s = 0; s < KSTEPS; ++s) {
    const int k0 = s * 32;
    if (tid < 32) {
      if (s + 1 < KSTEPS) {
        tdm_load_panel(lds0 + ((s + 1) & 1) * PANELB, wrow + k0 + 32, 3 * DM);
        __builtin_amdgcn_s_wait_tensorcnt(1);
      } else {
        __builtin_amdgcn_s_wait_tensorcnt(0);
      }
    }
    __syncthreads();
    v16h a_nxt;
    if (s + 1 < KSTEPS) a_nxt = loadA(arow + k0 + 32, hi);   /* in flight during WMMAs */
    const char* panel = smem + (s & 1) * PANELB;
    v16h bf[4];
    for (int j = 0; j < 4; ++j) bf[j] = ldsB(panel, 16 * j + llo, hi);  /* batch ds loads */
    for (int j = 0; j < 4; ++j) acc[j] = wmma_f16(a_cur, bf[j], acc[j]);
    __syncthreads();
    a_cur = a_nxt;
  }

  const int region = n0 >> 10;                        /* 0=Q 1=K 2=V (uniform per block) */
  for (int j = 0; j < 4; ++j) {
    const int e  = n0 + 16 * j + llo;
    const int el = e & (DM - 1);
    const int h  = el >> 6;
    const int dh = el & (DH - 1);
    for (int r = 0; r < 8; ++r) {
      const int mrow = m0 + r + 8 * hi;               /* D layout: lane = col n, row = vgpr+8*hi */
      const int b = mrow >> 11, s = mrow & (SEQ - 1);
      float v = acc[j][r];
      if (region < 2) {
        /* RoPE: partner (dh^1) sits in the adjacent lane of the D tile */
        float partner = __shfl_xor(v, 1, 32);
        float inv_freq = __expf(-ROPE_LN * (float)(dh >> 1));
        float ang = (float)s * inv_freq;
        float cs = __cosf(ang), sn = __sinf(ang);
        float rv = v * cs + ((dh & 1) ? partner : -partner) * sn;
        if (region == 0) rv *= QK_SCALE;
        _Float16* dst = (region == 0) ? qh : kh;
        dst[(((size_t)b * NH + h) * SEQ + s) * DH + dh] = (_Float16)rv;
      } else {
        vth[(((size_t)b * NH + h) * DH + dh) * SEQ + s] = (_Float16)v;
      }
    }
  }
}

/* ---------------- causal flash attention ----------------
 * One wave per (b,h, 16-query tile). S^T = K.Q^T so softmax rows (=queries) are
 * per-lane; P^T feeds WMMA2's B operand directly: O^T = V^T . P^T.
 * grid: (128, 64), block 32. Output oh[b, s, h*64+d] f16. */
__global__ __launch_bounds__(32)
void flash_attn_kernel(const _Float16* __restrict__ qh, const _Float16* __restrict__ kh,
                       const _Float16* __restrict__ vth, _Float16* __restrict__ oh)
{
  const int lane = threadIdx.x & 31;
  const int llo = lane & 15, hi = lane >> 4;
  const int q0 = blockIdx.x * 16;
  const int bh = blockIdx.y;
  const int b = bh >> 4, h = bh & 15;

  const _Float16* kbase = kh + (size_t)bh * SEQ * DH;
  const _Float16* vbase = vth + (size_t)bh * DH * SEQ;

  const _Float16* qrow = qh + ((size_t)bh * SEQ + (q0 + llo)) * DH;
  v16h bq0 = loadB(qrow + 0, hi);
  v16h bq1 = loadB(qrow + 32, hi);

  v8f ot[4] = {};                       /* O^T accum: 4 tiles cover d=0..63, lane = q */
  float m_run = NEG_INF, l_run = 0.f;
  const int qg = q0 + llo;

  for (int kv0 = 0; kv0 <= q0; kv0 += 32) {
    const int kr0 = kv0 + llo;
    int kr1 = kv0 + 16 + llo; if (kr1 > SEQ - 1) kr1 = SEQ - 1;   /* masked anyway */

    /* batch all loads for this chunk: 4 K fragments + 4 V^T fragments in flight */
    v16h ak00 = loadA(kbase + (size_t)kr0 * DH,      hi);
    v16h ak01 = loadA(kbase + (size_t)kr0 * DH + 32, hi);
    v16h ak10 = loadA(kbase + (size_t)kr1 * DH,      hi);
    v16h ak11 = loadA(kbase + (size_t)kr1 * DH + 32, hi);
    v16h av[4];
    for (int t = 0; t < 4; ++t)
      av[t] = loadA(vbase + (size_t)(16 * t + llo) * SEQ + kv0, hi);

    v8f st0 = {}, st1 = {};             /* S^T tiles: rows = kv, cols(lanes) = q */
    st0 = wmma_f16(ak00, bq0, st0);
    st0 = wmma_f16(ak01, bq1, st0);
    st1 = wmma_f16(ak10, bq0, st1);
    st1 = wmma_f16(ak11, bq1, st1);

    /* causal mask + per-query running max */
    float p0[8], p1[8];
    float mx = NEG_INF;
    for (int r = 0; r < 8; ++r) {
      int kvg0 = kv0 + r + 8 * hi;
      int kvg1 = kv0 + 16 + r + 8 * hi;
      p0[r] = (kvg0 <= qg) ? st0[r] : NEG_INF;
      p1[r] = (kvg1 <= qg) ? st1[r] : NEG_INF;
      mx = fmaxf(mx, fmaxf(p0[r], p1[r]));
    }
    mx = fmaxf(mx, __shfl_xor(mx, 16, 32));
    float m_new = fmaxf(m_run, mx);
    float corr = __expf(m_run - m_new);
    float ps = 0.f;
    for (int r = 0; r < 8; ++r) {
      p0[r] = __expf(p0[r] - m_new);
      p1[r] = __expf(p1[r] - m_new);
      ps += p0[r] + p1[r];
    }
    ps += __shfl_xor(ps, 16, 32);
    l_run = l_run * corr + ps;
    m_run = m_new;

    /* repack P^T into the B-fragment: lanes 0-15 need kv 0..15, lanes 16-31 kv 16..31 */
    FragU pb;
    for (int r = 0; r < 8; ++r) {
      float x0 = __shfl_xor(p0[r], 16, 32);
      float x1 = __shfl_xor(p1[r], 16, 32);
      pb.e[r]     = (_Float16)(hi ? x1 : p0[r]);
      pb.e[8 + r] = (_Float16)(hi ? p1[r] : x0);
    }

    for (int t = 0; t < 4; ++t) {
      v8f a = ot[t];
      for (int r = 0; r < 8; ++r) a[r] *= corr;
      ot[t] = wmma_f16(av[t], pb.v, a);
    }
  }

  const float inv_l = 1.f / l_run;
  _Float16* orow = oh + ((size_t)b * SEQ + qg) * DM + h * DH;
  for (int t = 0; t < 4; ++t) {
    v8h ov;
    for (int r = 0; r < 8; ++r) ov[r] = (_Float16)(ot[t][r] * inv_l);
    *(v8h*)(orow + 16 * t + 8 * hi) = ov;   /* contiguous 16B per lane */
  }
}

/* ---------------- output projection GEMM ----------------
 * out[m, n] = sum_d oh[m, d] * Wout[n, d], fp32 result.
 * Same TDM-staged block structure. grid: (128, 16), block 128. */
__global__ __launch_bounds__(128)
void out_proj_kernel(const _Float16* __restrict__ ah, const _Float16* __restrict__ wh,
                     float* __restrict__ out)
{
  __shared__ char smem[2 * PANELB];
  const int tid = threadIdx.x;
  const int lane = tid & 31;
  const int llo = lane & 15, hi = lane >> 4;
  const int m0 = blockIdx.x * 64 + (tid >> 5) * 16;
  const int n0 = blockIdx.y * 64;
  const unsigned lds0 = (unsigned)(size_t)smem;

  const _Float16* arow = ah + (size_t)(m0 + llo) * DM;
  const _Float16* wrow = wh + (size_t)n0 * DM;

  v8f acc[4] = {};
  if (tid < 32) tdm_load_panel(lds0, wrow, DM);

  v16h a_cur = loadA(arow, hi);
  for (int s = 0; s < KSTEPS; ++s) {
    const int k0 = s * 32;
    if (tid < 32) {
      if (s + 1 < KSTEPS) {
        tdm_load_panel(lds0 + ((s + 1) & 1) * PANELB, wrow + k0 + 32, DM);
        __builtin_amdgcn_s_wait_tensorcnt(1);
      } else {
        __builtin_amdgcn_s_wait_tensorcnt(0);
      }
    }
    __syncthreads();
    v16h a_nxt;
    if (s + 1 < KSTEPS) a_nxt = loadA(arow + k0 + 32, hi);
    const char* panel = smem + (s & 1) * PANELB;
    v16h bf[4];
    for (int j = 0; j < 4; ++j) bf[j] = ldsB(panel, 16 * j + llo, hi);
    for (int j = 0; j < 4; ++j) acc[j] = wmma_f16(a_cur, bf[j], acc[j]);
    __syncthreads();
    a_cur = a_nxt;
  }

  for (int j = 0; j < 4; ++j) {
    const int n = n0 + 16 * j + llo;
    for (int r = 0; r < 8; ++r) {
      const int m = m0 + r + 8 * hi;
      out[(size_t)m * DM + n] = acc[j][r];
    }
  }
}

extern "C" void kernel_launch(void* const* d_in, const int* in_sizes, int n_in,
                              void* d_out, int out_size, void* d_ws, size_t ws_size,
                              hipStream_t stream) {
  (void)in_sizes; (void)n_in; (void)out_size; (void)ws_size;
  const float* x    = (const float*)d_in[0];
  const float* Wqkv = (const float*)d_in[1];
  const float* Wout = (const float*)d_in[2];
  float* out = (float*)d_out;

  char* ws = (char*)d_ws;
  size_t off = 0;
  auto wsAlloc = [&](size_t bytes) -> char* {
    char* p = ws + off;
    off += (bytes + 255) & ~(size_t)255;
    return p;
  };
  const size_t nx  = (size_t)BATCH * SEQ * DM;   /* 8.4M  */
  const size_t nwq = (size_t)3 * DM * DM;        /* 3.1M  */
  const size_t nwo = (size_t)DM * DM;            /* 1.0M  */
  _Float16* xh  = (_Float16*)wsAlloc(nx * 2);
  _Float16* wqh = (_Float16*)wsAlloc(nwq * 2);
  _Float16* woh = (_Float16*)wsAlloc(nwo * 2);
  _Float16* qh  = (_Float16*)wsAlloc(nx * 2);
  _Float16* kh  = (_Float16*)wsAlloc(nx * 2 + 4096);   /* pad: causal diagonal over-read */
  _Float16* vth = (_Float16*)wsAlloc(nx * 2 + 4096);
  _Float16* oh  = (_Float16*)wsAlloc(nx * 2);

  cvt_f32_f16_kernel<<<dim3((int)((nx  + 255) / 256)), 256, 0, stream>>>(x,    xh,  (int)nx);
  cvt_f32_f16_kernel<<<dim3((int)((nwq + 255) / 256)), 256, 0, stream>>>(Wqkv, wqh, (int)nwq);
  cvt_f32_f16_kernel<<<dim3((int)((nwo + 255) / 256)), 256, 0, stream>>>(Wout, woh, (int)nwo);

  qkv_rope_kernel<<<dim3(BATCH * SEQ / 64, 3 * DM / 64), 128, 0, stream>>>(xh, wqh, qh, kh, vth);
  flash_attn_kernel<<<dim3(SEQ / 16, BATCH * NH), 32, 0, stream>>>(qh, kh, vth, oh);
  out_proj_kernel<<<dim3(BATCH * SEQ / 64, DM / 64), 128, 0, stream>>>(oh, woh, out);
}
